// FinalAttention_2233382994485
// MI455X (gfx1250) — matrix operand order repaired and checked by reading, not compiled, not run on hardware
//
#include <hip/hip_runtime.h>
#include <hip/hip_bf16.h>

// ---------------------------------------------------------------------------
// FinalAttention for MI455X (gfx1250): bf16 WMMA pipeline, bf16 intermediates
//   qkv(bf16)  = x @ qkv_w + qkv_b                  (WMMA GEMM, fp32 A)
//   attn(bf16) = sliding-window GQA flash attention (WMMA QK^T + PV)
//   out(fp32)  = attn @ out_w + out_b               (WMMA GEMM, bf16 A, async LDS)
// ---------------------------------------------------------------------------

typedef __attribute__((ext_vector_type(16))) __bf16 v16bf;
typedef __attribute__((ext_vector_type(8)))  __bf16 v8bf;
typedef __attribute__((ext_vector_type(4)))  __bf16 v4bf;
typedef __attribute__((ext_vector_type(8)))  float  v8f;
typedef __attribute__((ext_vector_type(4)))  float  f4;
typedef __attribute__((ext_vector_type(4)))  int    v4i;

// addrspace-qualified pointee typedefs for the async-copy builtin
typedef __attribute__((address_space(1))) v4i* gas_v4i_ptr;  // global
typedef __attribute__((address_space(3))) v4i* las_v4i_ptr;  // LDS

static constexpr int S_LEN    = 2048;
static constexpr int BATCH    = 2;
static constexpr int HID      = 2048;
static constexpr int N_KV     = 4;
static constexpr int N_HEAD   = 16;
static constexpr int HEAD_DIM = 128;
static constexpr int WINDOW   = 512;
static constexpr int QKV_COLS = HID + 2 * N_KV * HEAD_DIM;  // 3072
static constexpr int ROWS     = S_LEN * BATCH;              // 4096

#if __has_builtin(__builtin_amdgcn_global_load_async_to_lds_b128)
#define HAVE_ASYNC_LDS 1
#else
#define HAVE_ASYNC_LDS 0
#endif

// 16-byte global -> LDS copy. Async (ASYNCcnt) on gfx1250 toolchains that
// expose the builtin; plain vector copy otherwise.
static __device__ __forceinline__ void copy16_g2l(const void* g, void* l) {
#if HAVE_ASYNC_LDS
    __builtin_amdgcn_global_load_async_to_lds_b128(
        (gas_v4i_ptr)(const void*)g, (las_v4i_ptr)l, 0, 0);
#else
    *(v8bf*)l = *(const v8bf*)g;
#endif
}

static __device__ __forceinline__ void wait_async_copies() {
#if HAVE_ASYNC_LDS
#if __has_builtin(__builtin_amdgcn_s_wait_asynccnt)
    __builtin_amdgcn_s_wait_asynccnt(0);
#else
    asm volatile("s_wait_asynccnt 0x0" ::: "memory");
#endif
#endif
}

static __device__ __forceinline__ v16bf cat16(v8bf lo, v8bf hi) {
    return __builtin_shufflevector(lo, hi, 0, 1, 2, 3, 4, 5, 6, 7,
                                           8, 9, 10, 11, 12, 13, 14, 15);
}

static __device__ __forceinline__ v8f wmma_bf16(v16bf a, v16bf b, v8f c) {
    return __builtin_amdgcn_wmma_f32_16x16x32_bf16(
        false, a, false, b, (short)0, c, false, false);
}

static __device__ __forceinline__ v8f zero8() {
    v8f z;
#pragma unroll
    for (int i = 0; i < 8; ++i) z[i] = 0.0f;
    return z;
}

// ---------------------------------------------------------------------------
// GEMM: C[M,N] = A[M,K] @ W[K,N] + bias[N]
//   ABF16:  A is bf16 (staged with async-to-LDS), else fp32 (convert staging)
//   OBF16:  C is bf16, else fp32
// Block: 256 threads = 8 waves (4 M x 2 N). Block tile 128x128, K-slab 32.
// ---------------------------------------------------------------------------
template <bool ABF16, bool OBF16>
__global__ __launch_bounds__(256) void gemm_bias_wmma(
    const void* __restrict__ Av, const float* __restrict__ W,
    const float* __restrict__ bias, void* __restrict__ Cv,
    int M, int N, int K) {
    __shared__ __bf16 la[128][40];   // [m][k]  (stride 80B, 16B aligned)
    __shared__ __bf16 lb[128][40];   // [n][k]  (W transposed so k is contiguous)

    const int tid  = threadIdx.x;
    const int tm   = blockIdx.y * 128;
    const int tn   = blockIdx.x * 128;
    const int w    = tid >> 5;
    const int lane = tid & 31;
    const int half = lane >> 4;
    const int l15  = lane & 15;
    const int wm   = w >> 1;   // 0..3
    const int wn   = w & 1;    // 0..1

    v8f acc[2][4];
#pragma unroll
    for (int i = 0; i < 2; ++i)
#pragma unroll
        for (int j = 0; j < 4; ++j) acc[i][j] = zero8();

    for (int kt = 0; kt < K; kt += 32) {
        // ---- stage A tile: 128 rows x 32 k ----
        if (ABF16) {
            const __bf16* A = (const __bf16*)Av;
#pragma unroll
            for (int it = 0; it < 2; ++it) {
                int ci  = it * 256 + tid;       // 0..511 16B-chunks
                int row = ci >> 2;              // 4 chunks per 64B row
                int kc  = (ci & 3) * 8;
                copy16_g2l(&A[(size_t)(tm + row) * K + kt + kc], &la[row][kc]);
            }
        } else {
            const float* A = (const float*)Av;
            f4 fa[4];                           // batch loads, then convert
#pragma unroll
            for (int it = 0; it < 4; ++it) {
                int fi  = it * 256 + tid;       // 0..1023 float4s
                int row = fi >> 3;
                int kc  = (fi & 7) * 4;
                fa[it] = *(const f4*)&A[(size_t)(tm + row) * K + kt + kc];
            }
#pragma unroll
            for (int it = 0; it < 4; ++it) {
                int fi  = it * 256 + tid;
                int row = fi >> 3;
                int kc  = (fi & 7) * 4;
                v4bf h;
#pragma unroll
                for (int u = 0; u < 4; ++u) h[u] = (__bf16)fa[it][u];
                *(v4bf*)&la[row][kc] = h;
            }
        }
        // ---- stage W tile transposed: lb[n][k] ----
        {
            f4 fw[4];
#pragma unroll
            for (int it = 0; it < 4; ++it) {
                int fi   = it * 256 + tid;      // 0..1023 float4s
                int krow = fi >> 5;
                int nc   = (fi & 31) * 4;
                fw[it] = *(const f4*)&W[(size_t)(kt + krow) * N + tn + nc];
            }
#pragma unroll
            for (int it = 0; it < 4; ++it) {
                int fi   = it * 256 + tid;
                int krow = fi >> 5;
                int nc   = (fi & 31) * 4;
#pragma unroll
                for (int u = 0; u < 4; ++u) lb[nc + u][krow] = (__bf16)fw[it][u];
            }
        }
        if (kt + 32 < K) {
            const int es = ABF16 ? 2 : 4;
            __builtin_prefetch((const char*)Av + ((size_t)(tm + (tid >> 1)) * K + kt + 32 + (tid & 1) * 16) * es, 0, 1);
            __builtin_prefetch(&W[(size_t)(kt + 32 + (tid >> 3)) * N + tn + (tid & 7) * 16], 0, 1);
        }
        wait_async_copies();
        __syncthreads();

        // ---- fragments + WMMA ----
        v16bf af[2];
#pragma unroll
        for (int fm = 0; fm < 2; ++fm) {
            int row = wm * 32 + fm * 16 + l15;
            v8bf lo = *(const v8bf*)&la[row][half * 8];
            v8bf hi = *(const v8bf*)&la[row][16 + half * 8];
            af[fm] = cat16(lo, hi);
        }
#pragma unroll
        for (int fn = 0; fn < 4; ++fn) {
            int col = wn * 64 + fn * 16 + l15;
            v8bf lo = *(const v8bf*)&lb[col][half * 8];
            v8bf hi = *(const v8bf*)&lb[col][16 + half * 8];
            v16bf bf = cat16(lo, hi);
#pragma unroll
            for (int fm = 0; fm < 2; ++fm)
                acc[fm][fn] = wmma_bf16(af[fm], bf, acc[fm][fn]);
        }
        __syncthreads();
    }

    // ---- epilogue: C layout m = v + 8*half, n = lane&15 ----
#pragma unroll
    for (int fn = 0; fn < 4; ++fn) {
        int gn = tn + wn * 64 + fn * 16 + l15;
        float bv = bias[gn];
#pragma unroll
        for (int fm = 0; fm < 2; ++fm) {
#pragma unroll
            for (int v = 0; v < 8; ++v) {
                int gm = tm + wm * 32 + fm * 16 + v + half * 8;
                float r = acc[fm][fn][v] + bv;
                if (OBF16) ((__bf16*)Cv)[(size_t)gm * N + gn] = (__bf16)r;
                else       ((float*)Cv)[(size_t)gm * N + gn] = r;
            }
        }
    }
}

// ---------------------------------------------------------------------------
// Sliding-window GQA flash attention over bf16 qkv, bf16 output.
// Grid: x = q-tile (16), y = b*head (32). Block: 256 threads = 8 waves.
// ---------------------------------------------------------------------------
__global__ __launch_bounds__(256) void attn_wmma(
    const __bf16* __restrict__ qkv, __bf16* __restrict__ attn) {
    __shared__ __bf16 lk[64][136];      // [key][d]   (B-layout for QK^T)
    __shared__ __bf16 lv[128][72];      // [d][key]   (V^T: B-layout for PV)
    __shared__ __bf16 lp[8][16][72];    // per-wave P (A-matrix relayout)

    const int tid  = threadIdx.x;
    const int w    = tid >> 5;
    const int lane = tid & 31;
    const int half = lane >> 4;
    const int l15  = lane & 15;

    const int q0 = blockIdx.x * 128;
    const int bh = blockIdx.y;
    const int b  = bh & 1;
    const int h  = bh >> 1;
    const int g  = h >> 2;

    const int koff = HID + g * HEAD_DIM;
    const int voff = HID + N_KV * HEAD_DIM + g * HEAD_DIM;

    // ---- Q fragments: direct 16B bf16 loads, held in registers ----
    v16bf aq[4];
    {
        const int     qrow = q0 + w * 16 + l15;
        const __bf16* qp   = &qkv[((size_t)qrow * BATCH + b) * QKV_COLS + h * HEAD_DIM];
#pragma unroll
        for (int kc = 0; kc < 4; ++kc) {
            v8bf lo = *(const v8bf*)(qp + kc * 32 + half * 8);
            v8bf hi = *(const v8bf*)(qp + kc * 32 + 16 + half * 8);
            aq[kc] = cat16(lo, hi);
        }
    }

    float rmax[8], rsum[8];
    v8f   o[8];
#pragma unroll
    for (int v = 0; v < 8; ++v) { rmax[v] = -1e30f; rsum[v] = 0.0f; }
#pragma unroll
    for (int fd = 0; fd < 8; ++fd) o[fd] = zero8();

    const float scale = 0.088388347648318447f;  // 1/sqrt(128)

    int jlo = q0 - WINDOW;
    if (jlo < 0) jlo = 0;

    for (int j0 = jlo; j0 < q0 + 128; j0 += 64) {
        // ---- stage K block via async-to-LDS: lk[key][d], 64x128 bf16 ----
#pragma unroll
        for (int it = 0; it < 4; ++it) {
            int ci  = it * 256 + tid;       // 0..1023 16B-chunks
            int key = ci >> 4;              // 16 chunks per 256B row
            int dc  = (ci & 15) * 8;
            copy16_g2l(&qkv[((size_t)(j0 + key) * BATCH + b) * QKV_COLS + koff + dc],
                       &lk[key][dc]);
        }
        // ---- stage V block transposed: lv[d][key] (batched loads, scatter) ----
        {
            v8bf fv[4];
#pragma unroll
            for (int it = 0; it < 4; ++it) {
                int ci  = it * 256 + tid;
                int key = ci >> 4;
                int dc  = (ci & 15) * 8;
                fv[it] = *(const v8bf*)&qkv[((size_t)(j0 + key) * BATCH + b) * QKV_COLS + voff + dc];
            }
#pragma unroll
            for (int it = 0; it < 4; ++it) {
                int ci  = it * 256 + tid;
                int key = ci >> 4;
                int dc  = (ci & 15) * 8;
#pragma unroll
                for (int u = 0; u < 8; ++u) lv[dc + u][key] = fv[it][u];
            }
        }
        wait_async_copies();
        __syncthreads();

        // ---- scores: S = Q K^T over d=128 ----
        v8f sc[4];
#pragma unroll
        for (int fn = 0; fn < 4; ++fn) sc[fn] = zero8();
#pragma unroll
        for (int kc = 0; kc < 4; ++kc) {
#pragma unroll
            for (int fn = 0; fn < 4; ++fn) {
                int col = fn * 16 + l15;
                v8bf lo = *(const v8bf*)&lk[col][kc * 32 + half * 8];
                v8bf hi = *(const v8bf*)&lk[col][kc * 32 + 16 + half * 8];
                sc[fn] = wmma_bf16(aq[kc], cat16(lo, hi), sc[fn]);
            }
        }

        // ---- scale + sliding-window causal mask ----
#pragma unroll
        for (int fn = 0; fn < 4; ++fn) {
            int j = j0 + fn * 16 + l15;
#pragma unroll
            for (int v = 0; v < 8; ++v) {
                int i = q0 + w * 16 + v + half * 8;
                bool ok = (j <= i) && ((i - j) <= WINDOW);
                float s = sc[fn][v] * scale;
                sc[fn][v] = ok ? s : -1e30f;
            }
        }

        // ---- online softmax ----
#pragma unroll
        for (int v = 0; v < 8; ++v) {
            float mx = fmaxf(fmaxf(sc[0][v], sc[1][v]), fmaxf(sc[2][v], sc[3][v]));
#pragma unroll
            for (int d2 = 1; d2 < 16; d2 <<= 1) mx = fmaxf(mx, __shfl_xor(mx, d2, 32));
            float newm = fmaxf(rmax[v], mx);
            float ne = fmaxf(newm, -8e29f);
            float oe = fmaxf(rmax[v], -8e29f);
            float corr = __expf(oe - ne);
            float psum = 0.0f;
#pragma unroll
            for (int fn = 0; fn < 4; ++fn) {
                float p = __expf(sc[fn][v] - ne);
                psum += p;
                lp[w][v + half * 8][fn * 16 + l15] = (__bf16)p;
            }
#pragma unroll
            for (int d2 = 1; d2 < 16; d2 <<= 1) psum += __shfl_xor(psum, d2, 32);
            rsum[v] = rsum[v] * corr + psum;
            rmax[v] = newm;
#pragma unroll
            for (int fd = 0; fd < 8; ++fd) o[fd][v] *= corr;
        }

        // ---- O += P V ----
#pragma unroll
        for (int kc2 = 0; kc2 < 2; ++kc2) {
            v8bf plo = *(const v8bf*)&lp[w][l15][kc2 * 32 + half * 8];
            v8bf phi = *(const v8bf*)&lp[w][l15][kc2 * 32 + 16 + half * 8];
            v16bf ap = cat16(plo, phi);
#pragma unroll
            for (int fd = 0; fd < 8; ++fd) {
                int col = fd * 16 + l15;
                v8bf blo = *(const v8bf*)&lv[col][kc2 * 32 + half * 8];
                v8bf bhi = *(const v8bf*)&lv[col][kc2 * 32 + 16 + half * 8];
                o[fd] = wmma_bf16(ap, cat16(blo, bhi), o[fd]);
            }
        }
        __syncthreads();
    }

    // ---- normalize and write bf16 attn at [s][b][h*128 + d] ----
#pragma unroll
    for (int v = 0; v < 8; ++v) {
        float inv = 1.0f / rsum[v];
        int row = q0 + w * 16 + v + half * 8;
#pragma unroll
        for (int fd = 0; fd < 8; ++fd) {
            int dcol = fd * 16 + l15;
            attn[((size_t)row * BATCH + b) * HID + h * HEAD_DIM + dcol] =
                (__bf16)(o[fd][v] * inv);
        }
    }
}

// ---------------------------------------------------------------------------
extern "C" void kernel_launch(void* const* d_in, const int* in_sizes, int n_in,
                              void* d_out, int out_size, void* d_ws, size_t ws_size,
                              hipStream_t stream) {
    (void)in_sizes; (void)n_in; (void)out_size; (void)ws_size;
    const float* x     = (const float*)d_in[0];
    const float* qkv_w = (const float*)d_in[1];
    const float* qkv_b = (const float*)d_in[2];
    const float* out_w = (const float*)d_in[3];
    const float* out_b = (const float*)d_in[4];
    float*       out   = (float*)d_out;

    __bf16* qkvb  = (__bf16*)d_ws;                       // ROWS x 3072 bf16
    __bf16* attnb = qkvb + (size_t)ROWS * QKV_COLS;      // ROWS x 2048 bf16

    dim3 blk(256);
    gemm_bias_wmma<false, true><<<dim3(QKV_COLS / 128, ROWS / 128), blk, 0, stream>>>(
        x, qkv_w, qkv_b, qkvb, ROWS, QKV_COLS, HID);
    attn_wmma<<<dim3(S_LEN / 128, BATCH * N_HEAD), blk, 0, stream>>>(qkvb, attnb);
    gemm_bias_wmma<true, false><<<dim3(HID / 128, ROWS / 128), blk, 0, stream>>>(
        attnb, out_w, out_b, out, ROWS, HID, HID);
}
